// LSSModelV2_90074054132499
// MI455X (gfx1250) — matrix operand order repaired and checked by reading, not compiled, not run on hardware
//
#include <hip/hip_runtime.h>
#include <hip/hip_bf16.h>
#include <math.h>

typedef __attribute__((ext_vector_type(16))) _Float16 v16h;
typedef __attribute__((ext_vector_type(8)))  _Float16 v8h;
typedef __attribute__((ext_vector_type(8)))  float    v8f;

// ---------------------------------------------------------------------------
// Problem constants
// ---------------------------------------------------------------------------
#define BB 2
#define NCAM 6
#define BN 12           // B*N
#define CCH 64          // context channels
#define DD 40           // depth bins
#define FH 12
#define FW 33
#define P1 (FH*FW)      // 396 pixels per camera image
#define NX 100
#define NY 100
#define NZ 4
#define P2 (NX*NY)      // 10000 BEV pixels
#define BEVC 256        // NZ*CCH

// ---------------------------------------------------------------------------
// WMMA GEMM:  C[M,N] = act(scale[n]*(A@B) + bias[n])
//   A: activations fp32, logical [M,K]
//     AMODE 0: NCHW 1x1  -> A(m,k) = X[(m/P)*Cin*P + k*P + m%P]
//     AMODE 1: implicit im2col 3x3 pad1 over NCHW (H,W), K = Cin*9
//   B: weights fp32 row-major [N,K] (OIHW flattening)
//   C: NCHW: out[(m/P)*N*P + n*P + m%P]
//   ACT: 0 none, 1 SiLU, 2 ReLU
// Tile: 128x64, BK=32. 256 threads = 8 waves; each wave owns a 32x32 output
// (4 accumulators, 2 A-frags x 2 B-frags -> 4 WMMA / K-step). Global loads
// for step i+1 are register-staged before the WMMAs of step i.
// ---------------------------------------------------------------------------
#define TM 128
#define TN 64
#define TK 32
#define LDSA 40   // padded row stride (halfs): 80B rows, 16B-aligned vectors
#define LDSB 40

__device__ inline float act_apply(float v, int ACT) {
  if (ACT == 1) return v / (1.0f + __expf(-v));   // SiLU
  if (ACT == 2) return fmaxf(v, 0.0f);            // ReLU
  return v;
}

// 16x32 f16 A fragment (ISA 7.12.2): lane<16: K j<8 -> j, j>=8 -> 16+(j-8);
// lane>=16 adds +8. Rows contiguous in LDS -> two 16B ds_load_b128.
__device__ inline v16h frag_a_lds(const _Float16* As, int mbase, int lane) {
  const int m  = mbase + (lane & 15);
  const int ko = (lane >> 4);               // 0 or 1 (x8 halfs)
  const v8h* p = (const v8h*)(As + m * LDSA);
  const v8h lo = p[ko];
  const v8h hi = p[ko + 2];
  v16h a;
#pragma unroll
  for (int i = 0; i < 8; ++i) { a[i] = lo[i]; a[i + 8] = hi[i]; }
  return a;
}

// 32x16 f16 B fragment: N = lane%16, K j<8 -> j, j>=8 -> 16+(j-8); lane>=16 adds +8.
// Bs stored [n][k] -> two 8-half runs = two ds_load_b128.
__device__ inline v16h frag_b_lds(const _Float16* Bs, int nbase, int lane) {
  const int n  = nbase + (lane & 15);
  const int ko = (lane >> 4);               // 0 or 1 (x8 halfs)
  const v8h* p = (const v8h*)(Bs + n * LDSB);
  const v8h lo = p[ko];
  const v8h hi = p[ko + 2];
  v16h b;
#pragma unroll
  for (int i = 0; i < 8; ++i) { b[i] = lo[i]; b[i + 8] = hi[i]; }
  return b;
}

// Branch-free global A tile load: thread t stages 16 fp32 (row t>>1,
// k-chunk (t&1)*16). OOB rows clamp address + multiply by 0.
template <int AMODE>
__device__ inline void load_a_tile(float (&reg)[16], const float* __restrict__ A,
                                   int Cin, int H, int W, int P, int M,
                                   int m0, int k0, int t) {
  const int row = t >> 1;
  const int kc  = (t & 1) * 16;
  const int gm  = m0 + row;
  const int gmc = gm < M ? gm : M - 1;
  const float keep = (gm < M) ? 1.0f : 0.0f;
  const int img = gmc / P;
  const int pix = gmc - img * P;
  if (AMODE == 0) {
    const float* base = A + ((size_t)img * Cin + (k0 + kc)) * P + pix;
#pragma unroll
    for (int i = 0; i < 16; ++i) reg[i] = keep * base[(size_t)i * P];
  } else {
    const int y = pix / W;
    const int x = pix - y * W;
    const float* ib = A + (size_t)img * Cin * P;
#pragma unroll
    for (int i = 0; i < 16; ++i) {
      const int gk = k0 + kc + i;
      const int c  = gk / 9;
      const int rr = gk - c * 9;
      const int yy = y + rr / 3 - 1;
      const int xx = x + rr % 3 - 1;
      const bool ok = (yy >= 0) && (yy < H) && (xx >= 0) && (xx < W);
      const int yyc = min(max(yy, 0), H - 1);
      const int xxc = min(max(xx, 0), W - 1);
      const float v = ib[(size_t)c * P + yyc * W + xxc];
      reg[i] = ok ? keep * v : 0.0f;
    }
  }
}

// Branch-free global B tile load: thread t stages 8 contiguous fp32 of row
// t>>2 via two float4 loads (global_load_b128) + prefetch 2 slabs ahead.
__device__ inline void load_b_tile(float (&reg)[8], const float* __restrict__ Bw,
                                   int N, int K, int n0, int k0, int t) {
  const int n   = t >> 2;
  const int kc  = (t & 3) * 8;
  const int gn  = n0 + n;
  const int gnc = gn < N ? gn : N - 1;
  const float keep = (gn < N) ? 1.0f : 0.0f;
  const float* row = Bw + (size_t)gnc * K + k0 + kc;
  const float4 u0 = ((const float4*)row)[0];
  const float4 u1 = ((const float4*)row)[1];
  reg[0] = keep * u0.x; reg[1] = keep * u0.y;
  reg[2] = keep * u0.z; reg[3] = keep * u0.w;
  reg[4] = keep * u1.x; reg[5] = keep * u1.y;
  reg[6] = keep * u1.z; reg[7] = keep * u1.w;
  if (k0 + 2 * TK < K) __builtin_prefetch((const void*)(row + 2 * TK), 0, 1);
}

template <int AMODE, int ACT>
__global__ __launch_bounds__(256)
void gemm_wmma(const float* __restrict__ A, int Cin, int H, int W, int P,
               const float* __restrict__ Bw,
               float* __restrict__ Out,
               int M, int N, int K,
               const float* __restrict__ scale,
               const float* __restrict__ bias) {
  __shared__ _Float16 As[TM * LDSA];   // 10240 B
  __shared__ _Float16 Bs[TN * LDSB];   //  5120 B

  const int t    = threadIdx.x;
  const int lane = t & 31;
  const int wv   = t >> 5;             // 0..7
  const int mq   = wv & 3;             // 32-row quadrant
  const int nh   = wv >> 2;            // 32-col half
  const int m0   = blockIdx.x * TM;
  const int n0   = blockIdx.y * TN;

  v8f acc00 = {}, acc01 = {}, acc10 = {}, acc11 = {};

  float aReg[16];
  float bReg[8];
  load_a_tile<AMODE>(aReg, A, Cin, H, W, P, M, m0, 0, t);
  load_b_tile(bReg, Bw, N, K, n0, 0, t);

  for (int k0 = 0; k0 < K; k0 += TK) {
    __syncthreads();                   // LDS reads of prev step complete
    // ---- commit staged tiles to LDS as f16 (vector ds_store_b128) ----
    {
      const int row = t >> 1;
      const int kc  = (t & 1) * 16;
      v8h h0, h1;
#pragma unroll
      for (int i = 0; i < 8; ++i) {
        h0[i] = (_Float16)aReg[i];
        h1[i] = (_Float16)aReg[i + 8];
      }
      *(v8h*)(As + row * LDSA + kc)     = h0;
      *(v8h*)(As + row * LDSA + kc + 8) = h1;
    }
    {
      const int n  = t >> 2;
      const int kc = (t & 3) * 8;
      v8h h;
#pragma unroll
      for (int i = 0; i < 8; ++i) h[i] = (_Float16)bReg[i];
      *(v8h*)(Bs + n * LDSB + kc) = h;
    }
    __syncthreads();

    // ---- stage next K-slab (overlaps with WMMA below) ----
    if (k0 + TK < K) {
      load_a_tile<AMODE>(aReg, A, Cin, H, W, P, M, m0, k0 + TK, t);
      load_b_tile(bReg, Bw, N, K, n0, k0 + TK, t);
    }

    // ---- compute: 2 A-frags x 2 B-frags -> 4 WMMA ----
    const v16h a0 = frag_a_lds(As, 32 * mq, lane);
    const v16h a1 = frag_a_lds(As, 32 * mq + 16, lane);
    const v16h b0 = frag_b_lds(Bs, 32 * nh, lane);
    const v16h b1 = frag_b_lds(Bs, 32 * nh + 16, lane);
    acc00 = __builtin_amdgcn_wmma_f32_16x16x32_f16(false, a0, false, b0,
                                                   (short)0, acc00, false, false);
    acc01 = __builtin_amdgcn_wmma_f32_16x16x32_f16(false, a0, false, b1,
                                                   (short)0, acc01, false, false);
    acc10 = __builtin_amdgcn_wmma_f32_16x16x32_f16(false, a1, false, b0,
                                                   (short)0, acc10, false, false);
    acc11 = __builtin_amdgcn_wmma_f32_16x16x32_f16(false, a1, false, b1,
                                                   (short)0, acc11, false, false);
  }

  // ---- epilogue: scale/bias/act, scatter to NCHW ----
  const int mlo = (lane & 16) ? 8 : 0;
#pragma unroll
  for (int mi = 0; mi < 2; ++mi) {
#pragma unroll
    for (int ni = 0; ni < 2; ++ni) {
      const v8f acc = mi ? (ni ? acc11 : acc10) : (ni ? acc01 : acc00);
      const int gn = n0 + 32 * nh + 16 * ni + (lane & 15);
      if (gn < N) {
        float sc = scale ? scale[gn] : 1.0f;
        float bs = bias ? bias[gn] : 0.0f;
#pragma unroll
        for (int r = 0; r < 8; ++r) {
          const int gm = m0 + 32 * mq + 16 * mi + r + mlo;
          if (gm < M) {
            float v = act_apply(acc[r] * sc + bs, ACT);
            const int img = gm / P;
            const int pix = gm - img * P;
            Out[((size_t)img * N + gn) * P + pix] = v;
          }
        }
      }
    }
  }
}

// ---------------------------------------------------------------------------
// SE block helpers
// ---------------------------------------------------------------------------
__global__ void se_reduce_mean(const float* __restrict__ x,
                               float* __restrict__ mean, int P) {
  const int idx = blockIdx.x;                // (img*C + c)
  const float* p = x + (size_t)idx * P;
  float s = 0.0f;
  for (int i = threadIdx.x; i < P; i += blockDim.x) s += p[i];
  __shared__ float red[256];
  red[threadIdx.x] = s;
  __syncthreads();
  for (int off = 128; off > 0; off >>= 1) {
    if (threadIdx.x < off) red[threadIdx.x] += red[threadIdx.x + off];
    __syncthreads();
  }
  if (threadIdx.x == 0) mean[idx] = red[0] / (float)P;
}

__global__ void se_fc(const float* __restrict__ mean,
                      const float* __restrict__ w1, const float* __restrict__ b1,
                      const float* __restrict__ w2, const float* __restrict__ b2,
                      float* __restrict__ gate, int Cin, int Hid) {
  const int img = blockIdx.x;
  __shared__ float sm[256];
  __shared__ float hid[32];
  for (int i = threadIdx.x; i < Cin; i += blockDim.x)
    sm[i] = mean[(size_t)img * Cin + i];
  __syncthreads();
  if (threadIdx.x < (unsigned)Hid) {
    float s = b1[threadIdx.x];
    const float* wr = w1 + (size_t)threadIdx.x * Cin;
    for (int c = 0; c < Cin; ++c) s += sm[c] * wr[c];
    hid[threadIdx.x] = s / (1.0f + __expf(-s));   // SiLU
  }
  __syncthreads();
  for (int n = threadIdx.x; n < Cin; n += blockDim.x) {
    float s = b2[n];
    const float* wr = w2 + (size_t)n * Hid;
    for (int h = 0; h < Hid; ++h) s += hid[h] * wr[h];
    gate[(size_t)img * Cin + n] = 1.0f / (1.0f + __expf(-s));  // sigmoid
  }
}

__global__ void se_scale(float* __restrict__ x, const float* __restrict__ gate,
                         int P, int n) {
  const int i = blockIdx.x * blockDim.x + threadIdx.x;
  if (i < n) x[i] *= gate[i / P];
}

// ---------------------------------------------------------------------------
// depth softmax over 40 bins (per camera pixel, in place)
// ---------------------------------------------------------------------------
__global__ void depth_softmax(float* __restrict__ d, int Mpix) {
  const int m = blockIdx.x * blockDim.x + threadIdx.x;
  if (m >= Mpix) return;
  const int img = m / P1;
  const int pix = m - img * P1;
  float* base = d + (size_t)img * DD * P1 + pix;
  float mx = -1e30f;
#pragma unroll 8
  for (int i = 0; i < DD; ++i) mx = fmaxf(mx, base[i * P1]);
  float s = 0.0f;
#pragma unroll 8
  for (int i = 0; i < DD; ++i) {
    const float e = __expf(base[i * P1] - mx);
    base[i * P1] = e;
    s += e;
  }
  const float inv = 1.0f / s;
#pragma unroll 8
  for (int i = 0; i < DD; ++i) base[i * P1] *= inv;
}

// ---------------------------------------------------------------------------
// voxel splat: one thread per frustum point; ctx (64ch) x depth outer product
// scattered into BEV layout bev[b, z*64+c, x*100+y] with float atomics.
// ---------------------------------------------------------------------------
__global__ void splat(const float* __restrict__ depth,
                      const float* __restrict__ ctx,
                      const float* __restrict__ rots,
                      const float* __restrict__ trans,
                      const float* __restrict__ intr,
                      float* __restrict__ bev) {
  const int p = blockIdx.x * blockDim.x + threadIdx.x;
  const int NPTS = BN * DD * P1;
  if (p >= NPTS) return;
  const int img = p / (DD * P1);
  int rem = p - img * (DD * P1);
  const int id  = rem / P1;
  const int pix = rem - id * P1;
  const int iy  = pix / FW;
  const int ix  = pix - iy * FW;

  const float px = (float)ix * (1055.0f / 32.0f);   // linspace(0,1055,33)
  const float py = (float)iy * (383.0f / 11.0f);    // linspace(0,383,12)
  const float dv = 4.0f + (float)id;                // arange(4,44)

  const float* K = intr + (size_t)img * 9;
  const float fx = K[0], cx = K[2], fy = K[4], cy = K[5];
  const float camx = (px - cx) * dv / fx;
  const float camy = (py - cy) * dv / fy;
  const float camz = dv;

  const float* R = rots + (size_t)img * 9;
  const float* T = trans + (size_t)img * 3;
  const float gx = R[0] * camx + R[1] * camy + R[2] * camz + T[0];
  const float gy = R[3] * camx + R[4] * camy + R[5] * camz + T[1];
  const float gz = R[6] * camx + R[7] * camy + R[8] * camz + T[2];

  const int vx = (int)floorf(gx + 50.0f);
  const int vy = (int)floorf(gy + 50.0f);
  const int vz = (int)floorf((gz + 10.0f) * 0.2f);
  if (vx < 0 || vx >= NX || vy < 0 || vy >= NY || vz < 0 || vz >= NZ) return;

  const float dw = depth[(size_t)img * DD * P1 + (size_t)id * P1 + pix];
  const int b = img / NCAM;
  float* dst = bev + (size_t)b * BEVC * P2 + (size_t)vz * CCH * P2 +
               vx * NY + vy;
  const float* csrc = ctx + (size_t)img * CCH * P1 + pix;
#pragma unroll 4
  for (int c = 0; c < CCH; ++c)
    atomicAdd(dst + (size_t)c * P2, csrc[(size_t)c * P1] * dw);
}

// ---------------------------------------------------------------------------
// elementwise utilities
// ---------------------------------------------------------------------------
__global__ void zero_f32(float* __restrict__ p, int n) {
  const int i = blockIdx.x * blockDim.x + threadIdx.x;
  if (i < n) p[i] = 0.0f;
}
__global__ void add_f32(float* __restrict__ a, const float* __restrict__ b,
                        int n) {
  const int i = blockIdx.x * blockDim.x + threadIdx.x;
  if (i < n) a[i] += b[i];
}

// ---------------------------------------------------------------------------
// launcher
// ---------------------------------------------------------------------------
extern "C" void kernel_launch(void* const* d_in, const int* in_sizes, int n_in,
                              void* d_out, int out_size, void* d_ws,
                              size_t ws_size, hipStream_t stream) {
  const float* feats = (const float*)d_in[0];
  const float* rots  = (const float*)d_in[1];
  const float* trans = (const float*)d_in[2];
  const float* intr  = (const float*)d_in[3];
  const float* nw1   = (const float*)d_in[4];
  const float* bn1s  = (const float*)d_in[5];
  const float* bn1b  = (const float*)d_in[6];
  const float* se1w1 = (const float*)d_in[7];
  const float* se1b1 = (const float*)d_in[8];
  const float* se1w2 = (const float*)d_in[9];
  const float* se1b2 = (const float*)d_in[10];
  const float* nw2   = (const float*)d_in[11];
  const float* bn2s  = (const float*)d_in[12];
  const float* bn2b  = (const float*)d_in[13];
  const float* dhw   = (const float*)d_in[14];
  const float* dhb   = (const float*)d_in[15];
  const float* chw   = (const float*)d_in[16];
  const float* chb   = (const float*)d_in[17];
  const float* bcw1  = (const float*)d_in[18];
  const float* bcs1  = (const float*)d_in[19];
  const float* bcb1  = (const float*)d_in[20];
  const float* bcsw1 = (const float*)d_in[21];
  const float* bcsb1 = (const float*)d_in[22];
  const float* bcsw2 = (const float*)d_in[23];
  const float* bcsb2 = (const float*)d_in[24];
  const float* bcw2  = (const float*)d_in[25];
  const float* bcs2  = (const float*)d_in[26];
  const float* bcb2  = (const float*)d_in[27];
  const float* dw1   = (const float*)d_in[28];
  const float* ds1   = (const float*)d_in[29];
  const float* db1   = (const float*)d_in[30];
  const float* dw2   = (const float*)d_in[31];
  const float* db2   = (const float*)d_in[32];
  float* out = (float*)d_out;

  // workspace carve-up (floats)
  float* ws    = (float*)d_ws;
  float* x1    = ws;                        // 12*256*396
  float* x2    = x1 + (size_t)BN * 256 * P1;
  float* depth = x2 + (size_t)BN * 256 * P1;     // 12*40*396
  float* ctx   = depth + (size_t)BN * DD * P1;   // 12*64*396
  float* bev   = ctx + (size_t)BN * CCH * P1;    // 2*256*10000
  float* h1    = bev + (size_t)BB * BEVC * P2;
  float* h2    = h1 + (size_t)BB * BEVC * P2;
  float* hd    = h2 + (size_t)BB * BEVC * P2;    // 2*128*10000
  float* mean1 = hd + (size_t)BB * 128 * P2;     // 12*256
  float* gate1 = mean1 + BN * 256;
  float* mean2 = gate1 + BN * 256;               // 2*256
  float* gate2 = mean2 + BB * 256;

  auto cdiv = [](int a, int b) { return (a + b - 1) / b; };
  const dim3 blk(256);

  const int M1 = BN * P1;    // 4752
  const int M2 = BB * P2;    // 20000

  // ---- camera net ----
  gemm_wmma<0, 1><<<dim3(cdiv(M1, TM), cdiv(256, TN)), blk, 0, stream>>>(
      feats, 1280, FH, FW, P1, nw1, x1, M1, 256, 1280, bn1s, bn1b);
  se_reduce_mean<<<BN * 256, 256, 0, stream>>>(x1, mean1, P1);
  se_fc<<<BN, 256, 0, stream>>>(mean1, se1w1, se1b1, se1w2, se1b2, gate1, 256, 32);
  se_scale<<<cdiv(BN * 256 * P1, 256), blk, 0, stream>>>(x1, gate1, P1,
                                                         BN * 256 * P1);
  gemm_wmma<1, 1><<<dim3(cdiv(M1, TM), cdiv(256, TN)), blk, 0, stream>>>(
      x1, 256, FH, FW, P1, nw2, x2, M1, 256, 2304, bn2s, bn2b);
  gemm_wmma<0, 0><<<dim3(cdiv(M1, TM), 1), blk, 0, stream>>>(
      x2, 256, FH, FW, P1, dhw, depth, M1, DD, 256, nullptr, dhb);
  depth_softmax<<<cdiv(M1, 256), blk, 0, stream>>>(depth, M1);
  gemm_wmma<0, 0><<<dim3(cdiv(M1, TM), 1), blk, 0, stream>>>(
      x2, 256, FH, FW, P1, chw, ctx, M1, CCH, 256, nullptr, chb);

  // ---- lift-splat into BEV ----
  const int nbev = BB * BEVC * P2;
  zero_f32<<<cdiv(nbev, 256), blk, 0, stream>>>(bev, nbev);
  splat<<<cdiv(BN * DD * P1, 256), blk, 0, stream>>>(depth, ctx, rots, trans,
                                                     intr, bev);

  // ---- BEV encoder ----
  gemm_wmma<1, 2><<<dim3(cdiv(M2, TM), cdiv(256, TN)), blk, 0, stream>>>(
      bev, 256, NX, NY, P2, bcw1, h1, M2, 256, 2304, bcs1, bcb1);
  se_reduce_mean<<<BB * 256, 256, 0, stream>>>(h1, mean2, P2);
  se_fc<<<BB, 256, 0, stream>>>(mean2, bcsw1, bcsb1, bcsw2, bcsb2, gate2, 256, 16);
  se_scale<<<cdiv(nbev, 256), blk, 0, stream>>>(h1, gate2, P2, nbev);
  gemm_wmma<1, 2><<<dim3(cdiv(M2, TM), cdiv(256, TN)), blk, 0, stream>>>(
      h1, 256, NX, NY, P2, bcw2, h2, M2, 256, 2304, bcs2, bcb2);
  add_f32<<<cdiv(nbev, 256), blk, 0, stream>>>(h2, bev, nbev);  // residual

  // ---- decoder ----
  gemm_wmma<1, 2><<<dim3(cdiv(M2, TM), cdiv(128, TN)), blk, 0, stream>>>(
      h2, 256, NX, NY, P2, dw1, hd, M2, 128, 2304, ds1, db1);
  gemm_wmma<0, 0><<<dim3(cdiv(M2, TM), 1), blk, 0, stream>>>(
      hd, 128, NX, NY, P2, dw2, out, M2, 16, 128, nullptr, db2);
}
